// MSITB_3796751090402
// MI455X (gfx1250) — matrix operand order repaired and checked by reading, not compiled, not run on hardware
//
#include <hip/hip_runtime.h>
#include <math.h>

// ---------------------------------------------------------------------------
// MI455X (gfx1250, wave32) implementation of the 3-branch cross-attention +
// CBAM block. Heavy GEMMs (pointwise conv, q.k^T, p.v, proj, 3x3 conv as 9
// shifted GEMM taps) run on v_wmma_f32_16x16x32_bf16 with f32 accumulate.
// One wave owns one 16x16 output tile. Fragment layouts used here (same k order for A and B):
//   A: lanes 0-15 = rows M, K split by lane-half h=lane>>4:
//      elem i -> k = k0 + 16h + i (contiguous 16)
//   B: lane&15 = N column, elem i -> k = k0 + 16h + i (contiguous 16)
//   C/D: vgpr g -> m = g + 8h, n = lane&15
// A operands are pre-packed to bf16 (conv weights to [tap][M][K]) so fragment
// loads are 16B/32B vector loads. Strided activations are staged per-wave
// through LDS with b128 loads; the conv stages a 3x18 halo reused by 9 taps.
// ---------------------------------------------------------------------------

typedef __attribute__((ext_vector_type(16))) __bf16 v16bf;
typedef __attribute__((ext_vector_type(8)))  __bf16 v8bf;
typedef __attribute__((ext_vector_type(8)))  float  v8f;
typedef __attribute__((ext_vector_type(4)))  float  v4f;

#define HWC 4096           // 64*64 pixels
#define NB  8              // batch
#define KVC 448            // concat channels
#define EPSV 1e-5f

__device__ __forceinline__ v16bf cat8(v8bf lo, v8bf hi)
{
    return __builtin_shufflevector(lo, hi, 0, 1, 2, 3, 4, 5, 6, 7,
                                   8, 9, 10, 11, 12, 13, 14, 15);
}

// ---------------- weight packing: f32 -> bf16
__global__ void ms_pack(const float* __restrict__ src, __bf16* __restrict__ dst, long n)
{
    long i = (long)blockIdx.x * blockDim.x + threadIdx.x;
    if (i < n) dst[i] = (__bf16)src[i];
}

// conv weights [M,K,3,3] -> bf16 [tap][M][K]
__global__ void ms_pack9(const float* __restrict__ src, __bf16* __restrict__ dst, int M, int K)
{
    long i = (long)blockIdx.x * blockDim.x + threadIdx.x;
    long tot = (long)M * K * 9;
    if (i >= tot) return;
    int tap = (int)(i % 9);
    long mk = i / 9;
    dst[(long)tap * M * K + mk] = (__bf16)src[i];
}

// ---------------- per-pixel LayerNorm over channels (optionally concat of 3)
__global__ void ms_ln(const float* __restrict__ x0, const float* __restrict__ x1,
                      const float* __restrict__ x2, int c0, int c1, int c2,
                      const float* __restrict__ g, const float* __restrict__ be,
                      float* __restrict__ out, int Ctot)
{
    int idx = blockIdx.x * blockDim.x + threadIdx.x;       // pixel id
    if (idx >= NB * HWC) return;
    int b = idx >> 12, p = idx & (HWC - 1);
    auto ld = [&](int c) -> float {
        if (c < c0) return x0[(long)b * c0 * HWC + (long)c * HWC + p];
        c -= c0;
        if (c < c1) return x1[(long)b * c1 * HWC + (long)c * HWC + p];
        c -= c1;
        return x2[(long)b * c2 * HWC + (long)c * HWC + p];
    };
    float s = 0.f, ss = 0.f;
    for (int c = 0; c < Ctot; ++c) { float v = ld(c); s += v; ss += v * v; }
    float mu = s / Ctot;
    float var = ss / Ctot - mu * mu;
    float istd = rsqrtf(var + EPSV);
    for (int c = 0; c < Ctot; ++c) {
        float v = (ld(c) - mu) * istd;
        out[(long)b * Ctot * HWC + (long)c * HWC + p] = v * g[c] + be[c];
    }
}

// ---------------- depthwise 3x3 conv (+bias), zero padding
__global__ void ms_dw3(const float* __restrict__ x, const float* __restrict__ w,
                       const float* __restrict__ bias, float* __restrict__ out, int C)
{
    long i = (long)blockIdx.x * blockDim.x + threadIdx.x;
    long total = (long)NB * C * HWC;
    if (i >= total) return;
    int p = (int)(i & (HWC - 1));
    long bc = i >> 12;
    int c = (int)(bc % C);
    int y = p >> 6, xx0 = p & 63;
    const float* wc = w + c * 9;
    const float* src = x + (bc << 12);
    float acc = bias[c];
#pragma unroll
    for (int dy = -1; dy <= 1; ++dy) {
        int yy = y + dy;
        if (yy < 0 || yy > 63) continue;
#pragma unroll
        for (int dx = -1; dx <= 1; ++dx) {
            int xx = xx0 + dx;
            if (xx < 0 || xx > 63) continue;
            acc += wc[(dy + 1) * 3 + (dx + 1)] * src[yy * 64 + xx];
        }
    }
    out[i] = acc;
}

// ---------------- WMMA GEMM: out[b,m,p] = sum_k W[m,k]*X[b,k,p] (+bias[m]) (+resid)
// W pre-packed bf16; X f32 staged through per-wave LDS tile.
template <bool HB, bool HR>
__global__ __launch_bounds__(256) void ms_gemm_aw(
    const __bf16* __restrict__ W, const float* __restrict__ X,
    const float* __restrict__ bias, const float* __restrict__ resid,
    float* __restrict__ out, int M, int K)
{
    __shared__ float xs_[8][32][20];               // 20 KiB / block
    int lane = threadIdx.x, wv = threadIdx.y;
    int tilesM = M >> 4;
    const int tilesN = HWC >> 4;                   // 256
    long tilesPerB = (long)tilesM * tilesN;
    long tile = (long)blockIdx.x * 8 + wv;
    if (tile >= tilesPerB * NB) return;
    int b = (int)(tile / tilesPerB);
    long r = tile - (long)b * tilesPerB;
    int tm = (int)(r / tilesN), tn = (int)(r - (long)tm * tilesN);
    int m0 = tm << 4, n0 = tn << 4;
    int mn = lane & 15, h = lane >> 4;
    float (*xt)[20] = xs_[wv];
    const __bf16* Wrow = W + (long)(m0 + mn) * K;
    const float* Xb = X + (long)b * K * HWC + n0;
    v8f acc = {};
    for (int k0 = 0; k0 < K; k0 += 32) {
        // stage 32x16 B tile: lane owns K-row k0+lane (16 contiguous floats)
        const float* sr = Xb + (long)(k0 + lane) * HWC;
        __builtin_prefetch(sr + 32 * HWC, 0, 0);   // -> global_prefetch_b8
        v4f r0 = *(const v4f*)(sr + 0);
        v4f r1 = *(const v4f*)(sr + 4);
        v4f r2 = *(const v4f*)(sr + 8);
        v4f r3 = *(const v4f*)(sr + 12);
        *(v4f*)&xt[lane][0]  = r0;
        *(v4f*)&xt[lane][4]  = r1;
        *(v4f*)&xt[lane][8]  = r2;
        *(v4f*)&xt[lane][12] = r3;
        v16bf a = *(const v16bf*)(Wrow + k0 + 16 * h);
        v16bf bb;
#pragma unroll
        for (int i = 0; i < 16; ++i)
            bb[i] = (__bf16)xt[i + 16 * h][mn];
        acc = __builtin_amdgcn_wmma_f32_16x16x32_bf16(false, a, false, bb,
                                                      (short)0, acc, false, false);
    }
#pragma unroll
    for (int g = 0; g < 8; ++g) {
        int m = m0 + g + 8 * h;
        float v = acc[g];
        if (HB) v += bias[m];
        long o = (long)b * M * HWC + (long)m * HWC + n0 + mn;
        if (HR) v += resid[o];
        out[o] = v;
    }
}

// ---------------- WMMA NT GEMM: S[b,m,n] = scale * sum_p Q[b,m,p]*Kt[b,n,p]
__global__ __launch_bounds__(256) void ms_gemm_nt(
    const __bf16* __restrict__ Q, const __bf16* __restrict__ Kt,
    float* __restrict__ S, int M, float scale)
{
    int lane = threadIdx.x, wv = threadIdx.y;
    int tilesM = M >> 4;
    const int tilesN = KVC >> 4;                   // 28
    long tilesPerB = (long)tilesM * tilesN;
    long tile = (long)blockIdx.x * 8 + wv;
    if (tile >= tilesPerB * NB) return;
    int b = (int)(tile / tilesPerB);
    long r = tile - (long)b * tilesPerB;
    int tm = (int)(r / tilesN), tn = (int)(r % tilesN);
    int m0 = tm << 4, n0 = tn << 4;
    int mn = lane & 15, h = lane >> 4;
    const __bf16* Ar = Q  + ((long)b * M   + m0 + mn) * HWC;
    const __bf16* Br = Kt + ((long)b * KVC + n0 + mn) * HWC;
    v8f acc = {};
#pragma unroll 2
    for (int k0 = 0; k0 < HWC; k0 += 32) {
        __builtin_prefetch(Ar + k0 + 256, 0, 0);
        __builtin_prefetch(Br + k0 + 256, 0, 0);
        v16bf a = *(const v16bf*)(Ar + k0 + 16 * h);
        v16bf bb = *(const v16bf*)(Br + k0 + 16 * h);
        acc = __builtin_amdgcn_wmma_f32_16x16x32_bf16(false, a, false, bb,
                                                      (short)0, acc, false, false);
    }
#pragma unroll
    for (int g = 0; g < 8; ++g) {
        int m = m0 + g + 8 * h;
        S[((long)b * M + m) * KVC + n0 + mn] = acc[g] * scale;
    }
}

// ---------------- WMMA NN GEMM: O[b,m,p] = sum_k P[b,m,k]*V[b,k,p] (bf16 ops)
__global__ __launch_bounds__(256) void ms_gemm_pv(
    const __bf16* __restrict__ P, const __bf16* __restrict__ V,
    float* __restrict__ O, int M)
{
    __shared__ __bf16 vt_[8][32][24];              // 12 KiB / block
    int lane = threadIdx.x, wv = threadIdx.y;
    int tilesM = M >> 4;
    const int tilesN = HWC >> 4;                   // 256
    long tilesPerB = (long)tilesM * tilesN;
    long tile = (long)blockIdx.x * 8 + wv;
    if (tile >= tilesPerB * NB) return;
    int b = (int)(tile / tilesPerB);
    long r = tile - (long)b * tilesPerB;
    int tm = (int)(r / tilesN), tn = (int)(r % tilesN);
    int m0 = tm << 4, n0 = tn << 4;
    int mn = lane & 15, h = lane >> 4;
    __bf16 (*vt)[24] = vt_[wv];
    const __bf16* Ar = P + ((long)b * M + m0 + mn) * KVC;
    const __bf16* Vb = V + (long)b * KVC * HWC + n0;
    v8f acc = {};
    for (int k0 = 0; k0 < KVC; k0 += 32) {
        const __bf16* sr = Vb + (long)(k0 + lane) * HWC;
        v8bf w0 = *(const v8bf*)(sr);
        v8bf w1 = *(const v8bf*)(sr + 8);
        *(v8bf*)&vt[lane][0] = w0;
        *(v8bf*)&vt[lane][8] = w1;
        v16bf a = *(const v16bf*)(Ar + k0 + 16 * h);
        v16bf bb;
#pragma unroll
        for (int i = 0; i < 16; ++i)
            bb[i] = vt[i + 16 * h][mn];
        acc = __builtin_amdgcn_wmma_f32_16x16x32_bf16(false, a, false, bb,
                                                      (short)0, acc, false, false);
    }
#pragma unroll
    for (int g = 0; g < 8; ++g) {
        int m = m0 + g + 8 * h;
        O[(long)b * M * HWC + (long)m * HWC + n0 + mn] = acc[g];
    }
}

// ---------------- WMMA 3x3 conv: halo staged once per K-chunk, 9 taps reuse it
// Wp pre-packed bf16 [tap][M][K]; epilogue fuses bias + BN (+ReLU).
__global__ __launch_bounds__(256) void ms_conv3(
    const float* __restrict__ X, const __bf16* __restrict__ Wp,
    const float* __restrict__ cb, const float* __restrict__ bng,
    const float* __restrict__ bnb, float* __restrict__ out,
    int M, int K, int relu)
{
    __shared__ float hx[8][32][60];                // 60 KiB / block: 3 rows x (16+2) halo
    int lane = threadIdx.x, wv = threadIdx.y;
    int tilesM = M >> 4;
    const int tilesN = HWC >> 4;                   // 256 (16-pixel row segments)
    long tilesPerB = (long)tilesM * tilesN;
    long tile = (long)blockIdx.x * 8 + wv;
    if (tile >= tilesPerB * NB) return;
    int b = (int)(tile / tilesPerB);
    long r = tile - (long)b * tilesPerB;
    int tm = (int)(r / tilesN), tn = (int)(r % tilesN);
    int m0 = tm << 4;
    int y = tn >> 2, xs = (tn & 3) << 4;
    int mn = lane & 15, h = lane >> 4;
    float (*ht)[60] = hx[wv];
    const float* Xk = X + (long)b * K * HWC;
    long MK = (long)M * K;
    v8f acc = {};
    for (int k0 = 0; k0 < K; k0 += 32) {
        // stage rows y-1..y+1, cols xs-1..xs+16 for K-rows k0..k0+31
        const float* sr = Xk + (long)(k0 + lane) * HWC;
#pragma unroll
        for (int ry = 0; ry < 3; ++ry) {
            int yy = y + ry - 1;
            bool rv = (yy >= 0) && (yy < 64);
            const float* rp = sr + yy * 64 + xs;
            v4f z = {0.f, 0.f, 0.f, 0.f};
            v4f q0 = rv ? *(const v4f*)(rp + 0)  : z;
            v4f q1 = rv ? *(const v4f*)(rp + 4)  : z;
            v4f q2 = rv ? *(const v4f*)(rp + 8)  : z;
            v4f q3 = rv ? *(const v4f*)(rp + 12) : z;
            *(v4f*)&ht[lane][ry * 20 + 0]  = q0;
            *(v4f*)&ht[lane][ry * 20 + 4]  = q1;
            *(v4f*)&ht[lane][ry * 20 + 8]  = q2;
            *(v4f*)&ht[lane][ry * 20 + 12] = q3;
            ht[lane][ry * 20 + 16] = (rv && (xs + 16 < 64)) ? rp[16] : 0.f;   // col xs+16
            ht[lane][ry * 20 + 17] = (rv && (xs > 0))       ? rp[-1] : 0.f;   // col xs-1
        }
#pragma unroll
        for (int tap = 0; tap < 9; ++tap) {
            int ry = tap / 3, dx = tap % 3 - 1;
            const __bf16* wr = Wp + (long)tap * MK + (long)(m0 + mn) * K;
            v16bf a = *(const v16bf*)(wr + k0 + 16 * h);
            int cc = mn + dx;
            cc = (cc < 0) ? 17 : cc;               // -1 -> slot 17; 16 -> slot 16
            v16bf bb;
#pragma unroll
            for (int i = 0; i < 16; ++i)
                bb[i] = (__bf16)ht[i + 16 * h][ry * 20 + cc];
            acc = __builtin_amdgcn_wmma_f32_16x16x32_bf16(false, a, false, bb,
                                                          (short)0, acc, false, false);
        }
    }
    const float bsc = rsqrtf(1.f + EPSV);          // BN eval: running stats 0/1
#pragma unroll
    for (int g = 0; g < 8; ++g) {
        int m = m0 + g + 8 * h;
        float sc = bng[m] * bsc;
        float v = (acc[g] + cb[m]) * sc + bnb[m];
        if (relu) v = fmaxf(v, 0.f);
        out[(long)b * M * HWC + (long)m * HWC + y * 64 + xs + mn] = v;
    }
}

// ---------------- per-row (b,c over HW) l2norm -> bf16 (or plain cast)
__global__ void ms_rownorm(const float* __restrict__ X, __bf16* __restrict__ Y, int normalize)
{
    long row = blockIdx.x;
    const float* x = X + row * HWC;
    __shared__ float red[256];
    int tid = threadIdx.x;
    float ss = 0.f;
    for (int p = tid; p < HWC; p += 256) { float v = x[p]; ss += v * v; }
    red[tid] = ss; __syncthreads();
    for (int s = 128; s > 0; s >>= 1) { if (tid < s) red[tid] += red[tid + s]; __syncthreads(); }
    float inv = normalize ? (1.f / fmaxf(sqrtf(red[0]), 1e-12f)) : 1.f;
    for (int p = tid; p < HWC; p += 256) Y[row * HWC + p] = (__bf16)(x[p] * inv);
}

// ---------------- instance-norm stats over (M x 448) per batch
__global__ void ms_instat(const float* __restrict__ S, float* __restrict__ st, int M)
{
    int b = blockIdx.x, tid = threadIdx.x;
    long N = (long)M * KVC;
    const float* s = S + (long)b * N;
    __shared__ float r1[256], r2[256];
    float su = 0.f, ss = 0.f;
    for (long i = tid; i < N; i += 256) { float v = s[i]; su += v; ss += v * v; }
    r1[tid] = su; r2[tid] = ss; __syncthreads();
    for (int k = 128; k > 0; k >>= 1) {
        if (tid < k) { r1[tid] += r1[tid + k]; r2[tid] += r2[tid + k]; }
        __syncthreads();
    }
    if (!tid) {
        float mu = r1[0] / (float)N;
        float var = r2[0] / (float)N - mu * mu;
        st[b * 2] = mu;
        st[b * 2 + 1] = rsqrtf(var + EPSV);
    }
}

// ---------------- row softmax of IN-normalized scores -> bf16 probs
__global__ void ms_softmax(const float* __restrict__ S, const float* __restrict__ st,
                           __bf16* __restrict__ P, int M)
{
    long row = blockIdx.x;
    int b = (int)(row / M);
    const float* s = S + row * KVC;
    float mu = st[b * 2], istd = st[b * 2 + 1];
    __shared__ float red[256];
    int tid = threadIdx.x;
    float mx = -1e30f;
    for (int i = tid; i < KVC; i += 256) mx = fmaxf(mx, (s[i] - mu) * istd);
    red[tid] = mx; __syncthreads();
    for (int k = 128; k > 0; k >>= 1) { if (tid < k) red[tid] = fmaxf(red[tid], red[tid + k]); __syncthreads(); }
    mx = red[0]; __syncthreads();
    float sum = 0.f;
    for (int i = tid; i < KVC; i += 256) sum += expf((s[i] - mu) * istd - mx);
    red[tid] = sum; __syncthreads();
    for (int k = 128; k > 0; k >>= 1) { if (tid < k) red[tid] += red[tid + k]; __syncthreads(); }
    float inv = 1.f / red[0];
    for (int i = tid; i < KVC; i += 256)
        P[row * KVC + i] = (__bf16)(expf((s[i] - mu) * istd - mx) * inv);
}

// ---------------- CBAM channel pooling (avg/max over HW per (b,c) row)
__global__ void ms_hwstats(const float* __restrict__ X, float* __restrict__ avg,
                           float* __restrict__ mx)
{
    long row = blockIdx.x;
    const float* x = X + row * HWC;
    __shared__ float r1[256], r2[256];
    int tid = threadIdx.x;
    float s = 0.f, m = -1e30f;
    for (int p = tid; p < HWC; p += 256) { float v = x[p]; s += v; m = fmaxf(m, v); }
    r1[tid] = s; r2[tid] = m; __syncthreads();
    for (int k = 128; k > 0; k >>= 1) {
        if (tid < k) { r1[tid] += r1[tid + k]; r2[tid] = fmaxf(r2[tid], r2[tid + k]); }
        __syncthreads();
    }
    if (!tid) { avg[row] = r1[0] * (1.f / HWC); mx[row] = r2[0]; }
}

// ---------------- CBAM channel MLP: gate = sigmoid(W2 relu(W1 avg) + W2 relu(W1 mx))
__global__ void ms_camlp(const float* __restrict__ avg, const float* __restrict__ mx,
                         const float* __restrict__ W1, const float* __restrict__ W2,
                         float* __restrict__ gate, int C, int rr)
{
    int b = blockIdx.x, tid = threadIdx.x;
    __shared__ float ha[16], hm[16];
    if (tid < rr) {
        float sa = 0.f, sm = 0.f;
        for (int c = 0; c < C; ++c) {
            float w = W1[tid * C + c];
            sa += w * avg[b * C + c];
            sm += w * mx[b * C + c];
        }
        ha[tid] = fmaxf(sa, 0.f);
        hm[tid] = fmaxf(sm, 0.f);
    }
    __syncthreads();
    for (int c = tid; c < C; c += blockDim.x) {
        float s = 0.f;
        for (int j = 0; j < rr; ++j) s += W2[c * rr + j] * (ha[j] + hm[j]);
        gate[b * C + c] = 1.f / (1.f + expf(-s));
    }
}

__global__ void ms_applyca(float* __restrict__ X, const float* __restrict__ gate, long total)
{
    long i = (long)blockIdx.x * blockDim.x + threadIdx.x;
    if (i >= total) return;
    X[i] *= gate[i >> 12];
}

// ---------------- spatial mean/max over channels -> SA[b,2,HW]
__global__ void ms_spstats(const float* __restrict__ X, float* __restrict__ SA, int C)
{
    int idx = blockIdx.x * blockDim.x + threadIdx.x;
    if (idx >= NB * HWC) return;
    int b = idx >> 12, p = idx & (HWC - 1);
    const float* x = X + (long)b * C * HWC + p;
    float s = 0.f, m = -1e30f;
    for (int c = 0; c < C; ++c) { float v = x[(long)c * HWC]; s += v; m = fmaxf(m, v); }
    SA[(long)b * 2 * HWC + p] = s / (float)C;
    SA[(long)b * 2 * HWC + HWC + p] = m;
}

// ---------------- spatial 7x7 conv on 2 channels -> sigmoid gate per pixel
__global__ void ms_saconv(const float* __restrict__ SA, const float* __restrict__ w,
                          float* __restrict__ gsp)
{
    int idx = blockIdx.x * blockDim.x + threadIdx.x;
    if (idx >= NB * HWC) return;
    int b = idx >> 12, p = idx & (HWC - 1);
    int y = p >> 6, x0 = p & 63;
    float acc = 0.f;
    for (int ci = 0; ci < 2; ++ci)
        for (int ky = 0; ky < 7; ++ky) {
            int yy = y + ky - 3;
            if (yy < 0 || yy > 63) continue;
            for (int kx = 0; kx < 7; ++kx) {
                int xx = x0 + kx - 3;
                if (xx < 0 || xx > 63) continue;
                acc += w[ci * 49 + ky * 7 + kx] * SA[(long)b * 2 * HWC + (long)ci * HWC + yy * 64 + xx];
            }
        }
    gsp[idx] = 1.f / (1.f + expf(-acc));
}

// ---------------- final: relu(F2*sa_gate + ln_in) + cx -> d_out
__global__ void ms_final(const float* __restrict__ F2, const float* __restrict__ gsp,
                         const float* __restrict__ lnin, const float* __restrict__ cx,
                         float* __restrict__ out, int C)
{
    long i = (long)blockIdx.x * blockDim.x + threadIdx.x;
    long total = (long)NB * C * HWC;
    if (i >= total) return;
    int p = (int)(i & (HWC - 1));
    int b = (int)(i / ((long)C * HWC));
    float v = F2[i] * gsp[(long)b * HWC + p] + lnin[i];
    out[i] = fmaxf(v, 0.f) + cx[i];
}

// ---------------------------------------------------------------------------
// Host orchestration. Params are the jax-pytree (sorted dict key) flattening:
//  0 emb1  1 emb2  2 emb3
//  3/4 an1(w,b) 5/6 an2 7/8 an3 9/10 ana
//  11..21 ffn1: bn1b bn1g bn2b bn2g c1b c1w c2b c2w caw1 caw2 saw
//  22..32 ffn2, 33..43 ffn3
//  44/45 fn1 46/47 fn2 48/49 fn3
//  50..53 k: dw dwb pw pwb
//  54 proj1 55 proj2 56 proj3
//  57..60 q1, 61..64 q2, 65..68 q3
//  69..72 v: dw dwb pw pwb
// ---------------------------------------------------------------------------
extern "C" void kernel_launch(void* const* d_in, const int* in_sizes, int n_in,
                              void* d_out, int out_size, void* d_ws, size_t ws_size,
                              hipStream_t stream)
{
    (void)in_sizes; (void)n_in; (void)out_size; (void)ws_size;
    const float* e1 = (const float*)d_in[0];
    const float* e2 = (const float*)d_in[1];
    const float* e3 = (const float*)d_in[2];
    auto P = [&](int i) { return (const float*)d_in[i]; };

    const long SZ448 = (long)NB * KVC * HWC;   // 14,680,064
    const long SZ256 = (long)NB * 256 * HWC;   //  8,388,608
    float*  ws  = (float*)d_ws;
    float*  LNA = ws;                          // LN(concat); later reused as F2
    float*  TMP = LNA + SZ448;                 // depthwise scratch
    float*  PW  = TMP + SZ448;                 // pointwise f32 out; later F1
    __bf16* KB  = (__bf16*)(PW + SZ448);       // k (bf16, l2-normalized)
    __bf16* VB  = KB + SZ448;                  // v (bf16)
    __bf16* QB  = (__bf16*)(VB + SZ448);       // q (bf16), max 256 ch
    float*  LN1 = (float*)(QB + SZ256);        // per-branch LN buffer
    float*  Sb  = LN1 + SZ256;                 // attn scores (b, C, 448)
    __bf16* Pb  = (__bf16*)(Sb + (long)NB * 256 * KVC); // softmax probs bf16
    float*  Ob  = (float*)(Pb + (long)NB * 256 * KVC);  // attn out / pw out
    float*  CX  = Ob + SZ256;                  // cx1|cx2|cx3 (448-ch worth)
    float*  AVG = CX + SZ448;
    float*  MXX = AVG + NB * 256;
    float*  GATE= MXX + NB * 256;
    float*  SA  = GATE + NB * 256;
    float*  GSP = SA + (long)NB * 2 * HWC;
    float*  ST  = GSP + (long)NB * HWC;
    // packed bf16 weights
    __bf16* WK  = (__bf16*)(ST + 64);          // k pw   448*448
    __bf16* WV  = WK + 200704;                 // v pw   448*448
    __bf16* WQ  = WV + 200704;                 // q pw   <= 256*256
    __bf16* WJ  = WQ + 65536;                  // proj   <= 256*256
    __bf16* WC1 = WJ + 65536;                  // conv1  <= 9*256*256
    __bf16* WC2 = WC1 + 589824;                // conv2  <= 9*256*256
    float*  F1  = PW;                          // free after k/v built
    float*  F2  = LNA;                         // free after k/v built

    dim3 wblk(32, 8);                          // 8 waves/block, 1 tile/wave
    auto gemmBlocks = [](int M, int tilesN) {
        long w = (long)(M >> 4) * tilesN * NB;
        return (int)((w + 7) / 8);
    };
    auto cdiv = [](long a, long b) { return (int)((a + b - 1) / b); };

    // LN over concatenated channels -> LNA
    ms_ln<<<128, 256, 0, stream>>>(e1, e2, e3, 64, 128, 256, P(9), P(10), LNA, KVC);
    // pack k/v pointwise weights to bf16
    ms_pack<<<cdiv(200704, 256), 256, 0, stream>>>(P(52), WK, 200704);
    ms_pack<<<cdiv(200704, 256), 256, 0, stream>>>(P(71), WV, 200704);
    // k = l2norm(dsconv(LNA))
    ms_dw3<<<cdiv(SZ448, 256), 256, 0, stream>>>(LNA, P(50), P(51), TMP, KVC);
    ms_gemm_aw<true, false><<<gemmBlocks(KVC, 256), wblk, 0, stream>>>(
        WK, TMP, P(53), nullptr, PW, KVC, KVC);
    ms_rownorm<<<NB * KVC, 256, 0, stream>>>(PW, KB, 1);
    // v = dsconv(LNA)
    ms_dw3<<<cdiv(SZ448, 256), 256, 0, stream>>>(LNA, P(69), P(70), TMP, KVC);
    ms_gemm_aw<true, false><<<gemmBlocks(KVC, 256), wblk, 0, stream>>>(
        WV, TMP, P(72), nullptr, PW, KVC, KVC);
    ms_rownorm<<<NB * KVC, 256, 0, stream>>>(PW, VB, 0);

    const float scale = 1.f / sqrtf((float)KVC);
    const int   Cs[3]  = {64, 128, 256};
    const int   rs[3]  = {4, 8, 16};
    const int   anW[3] = {3, 5, 7};
    const int   qB[3]  = {57, 61, 65};
    const int   prj[3] = {54, 55, 56};
    const int   fnW[3] = {44, 46, 48};
    const int   ffB[3] = {11, 22, 33};
    const float* embs[3] = {e1, e2, e3};
    const long  off3[3] = {0, (long)NB * 64 * HWC, (long)NB * (64 + 128) * HWC};

    for (int i = 0; i < 3; ++i) {
        int  C = Cs[i];
        long szC = (long)NB * C * HWC;
        int  eltBlocks = cdiv(szC, 256);
        float* cx = CX + off3[i];
        // pack this branch's weights
        ms_pack<<<cdiv((long)C * C, 256), 256, 0, stream>>>(P(qB[i] + 2), WQ, (long)C * C);
        ms_pack<<<cdiv((long)C * C, 256), 256, 0, stream>>>(P(prj[i]), WJ, (long)C * C);
        ms_pack9<<<cdiv((long)C * C * 9, 256), 256, 0, stream>>>(P(ffB[i] + 5), WC1, C, C);
        ms_pack9<<<cdiv((long)C * C * 9, 256), 256, 0, stream>>>(P(ffB[i] + 7), WC2, C, C);
        // q = l2norm(dsconv(LN(emb)))
        ms_ln<<<128, 256, 0, stream>>>(embs[i], embs[i], embs[i], C, 0, 0,
                                       P(anW[i]), P(anW[i] + 1), LN1, C);
        ms_dw3<<<eltBlocks, 256, 0, stream>>>(LN1, P(qB[i]), P(qB[i] + 1), TMP, C);
        ms_gemm_aw<true, false><<<gemmBlocks(C, 256), wblk, 0, stream>>>(
            WQ, TMP, P(qB[i] + 3), nullptr, Ob, C, C);
        ms_rownorm<<<NB * C, 256, 0, stream>>>(Ob, QB, 1);
        // attention
        ms_gemm_nt<<<gemmBlocks(C, KVC >> 4), wblk, 0, stream>>>(QB, KB, Sb, C, scale);
        ms_instat<<<NB, 256, 0, stream>>>(Sb, ST, C);
        ms_softmax<<<NB * C, 256, 0, stream>>>(Sb, ST, Pb, C);
        ms_gemm_pv<<<gemmBlocks(C, 256), wblk, 0, stream>>>(Pb, VB, Ob, C);
        // cx = emb + proj(O)
        ms_gemm_aw<false, true><<<gemmBlocks(C, 256), wblk, 0, stream>>>(
            WJ, Ob, nullptr, embs[i], cx, C, C);
        // FFN: res_cbam(LN(cx)) + cx
        ms_ln<<<128, 256, 0, stream>>>(cx, cx, cx, C, 0, 0, P(fnW[i]), P(fnW[i] + 1), LN1, C);
        ms_conv3<<<gemmBlocks(C, 256), wblk, 0, stream>>>(
            LN1, WC1, P(ffB[i] + 4), P(ffB[i] + 1), P(ffB[i] + 0), F1, C, C, 1);
        ms_conv3<<<gemmBlocks(C, 256), wblk, 0, stream>>>(
            F1, WC2, P(ffB[i] + 6), P(ffB[i] + 3), P(ffB[i] + 2), F2, C, C, 0);
        ms_hwstats<<<NB * C, 256, 0, stream>>>(F2, AVG, MXX);
        ms_camlp<<<NB, 256, 0, stream>>>(AVG, MXX, P(ffB[i] + 8), P(ffB[i] + 9), GATE, C, rs[i]);
        ms_applyca<<<eltBlocks, 256, 0, stream>>>(F2, GATE, szC);
        ms_spstats<<<128, 256, 0, stream>>>(F2, SA, C);
        ms_saconv<<<128, 256, 0, stream>>>(SA, P(ffB[i] + 10), GSP);
        ms_final<<<eltBlocks, 256, 0, stream>>>(F2, GSP, LN1, cx,
                                                (float*)d_out + off3[i], C);
    }
}
